// FASTMultiHeadAttention_24043226923584
// MI455X (gfx1250) — matrix operand order repaired and checked by reading, not compiled, not run on hardware
//
#include <hip/hip_runtime.h>

// ---------------------------------------------------------------------------
// FAST (p=2 Taylor) linear attention for MI455X (gfx1250, wave32, WMMA).
// Roofline: 17.2 GFLOP vs 67 MB HBM (2.9us @ 23.3 TB/s) -> bf16 WMMA
// (v_wmma_f32_16x16x32_bf16, K=32==D) is the only precision that makes this
// memory-bound.  bf16 operands, f32 accumulation.
//
// Round-3: phase-2 LDS staging now uses CDNA5 async global->LDS DMA
// (global_load_async_to_lds_b128 + s_wait_asynccnt) instead of a VGPR
// round-trip copy.
// ---------------------------------------------------------------------------

typedef __attribute__((ext_vector_type(16))) __bf16 v16bf;
typedef __attribute__((ext_vector_type(8)))  float  v8f;

#define BB   2
#define HH   16
#define BHn  (BB*HH)          // 32 heads
#define NN   4096
#define DD   32
#define DD2  (DD*DD)          // 1024

#define kA0  1.0f
#define kA1  1.0f
#define kA2  0.5f

// ---- phase-1 split: private partial-stat images, no atomics ----
#define NSPLIT 4
#define NCHUNK (NN/NSPLIT)    // 1024 rows per block

// region A (f32 partial stats), per (head,split): [S2 | S1 | G | sk | sv]
#define STAT_F32 (DD2*DD + DD2 + DD2 + DD + DD)   // 34880 floats
#define S2_OFF 0
#define S1_OFF (DD2*DD)                  // 32768
#define G_OFF  (S1_OFF + DD2)            // 33792
#define SK_OFF (G_OFF  + DD2)            // 34816
#define SV_OFF (SK_OFF + DD)             // 34848
#define REGA_BYTES ((size_t)STAT_F32 * BHn * NSPLIT * 4)   // ~17.0 MiB

// region B: per-head bf16 B-operand images, column-major with padded strides
// (bank-skew), scales a1/a2 pre-folded.  Layout == phase-2 LDS layout so the
// LDS fill is a plain 16-byte-chunk DMA.
#define S2T_STRIDE   1026      // bf16 elems per e-row of S2^T (pad 2 -> skewed banks)
#define SMALL_STRIDE 34        // for S1^T and G^T
#define S2T_BYTE_OFF 0
#define S1T_BYTE_OFF (DD*S2T_STRIDE*2)                     // 65664
#define GT_BYTE_OFF  (S1T_BYTE_OFF + DD*SMALL_STRIDE*2)    // 67840
#define SVB_BYTE_OFF (GT_BYTE_OFF  + DD*SMALL_STRIDE*2)    // 70016 (f32)
#define SKB_BYTE_OFF (SVB_BYTE_OFF + DD*4)                 // 70144 (f32)
#define REGB_HEAD_BYTES (SKB_BYTE_OFF + DD*4)              // 70272 (16B multiple)
#define REGB_OFF REGA_BYTES
// total workspace ~= 17.0 MiB + 32*70272 B ~= 19.2 MiB

__device__ __forceinline__ v8f wmma_bf16(v16bf a, v16bf b, v8f c) {
  // v_wmma_f32_16x16x32_bf16  (neg_a, A, neg_b, B, c_mod, C, reuse_a, reuse_b)
  return __builtin_amdgcn_wmma_f32_16x16x32_bf16(false, a, false, b, (short)0, c,
                                                 false, false);
}

// CDNA5 async global->LDS copy of one 16B chunk (tracked by ASYNCcnt).
// ldsOff = wave-relative LDS byte offset, gaddr = 64-bit global address.
__device__ __forceinline__ void async_ld_b128(unsigned ldsOff,
                                              unsigned long long gaddr) {
  asm volatile("global_load_async_to_lds_b128 %0, %1, off"
               :: "v"(ldsOff), "v"(gaddr) : "memory");
}
__device__ __forceinline__ void wait_asynccnt0() {
  asm volatile("s_wait_asynccnt 0" ::: "memory");
}

// B-operand build from a transposed f32 LDS tile: B bf16 32x16 layout,
// this code's contiguous k order (same as A): half0 K=0..15 / half1 K=16..31, col = lane&15.
__device__ __forceinline__ v16bf buildB(const float* __restrict__ srcT,
                                        int col, int half) {
  v16bf b;
  #pragma unroll
  for (int i = 0; i < 16; ++i)
    b[i] = (__bf16)srcT[col*SMALL_STRIDE + half*16 + i];
  return b;
}

__device__ __forceinline__ v16bf buildBw(const float* __restrict__ srcT,
                                         const float* __restrict__ wRow,
                                         int col, int half) {
  v16bf b;
  #pragma unroll
  for (int i = 0; i < 16; ++i) {
    const int nn = half*16 + i;
    b[i] = (__bf16)(srcT[col*SMALL_STRIDE + nn] * wRow[nn]);
  }
  return b;
}

// ---------------------------------------------------------------------------
// Phase 1: per (head, N-split) partial stats via WMMA.
//   A = k^T tile (bf16, built once per 32-row chunk, shared by all segments).
//   Each wave owns 4 S2 d1-blocks:  C_{d1} += k^T @ (k[:,d1] (.) v).
//   wave0 also: S1 = k^T @ v.   wave1 also: G = k^T @ k  (reshape(sk2)).
//   wave2/3: sk / sv column sums (VALU).
//   All accumulators statically indexed -> VGPR-resident.
// ---------------------------------------------------------------------------
__global__ __launch_bounds__(256)
void fastmax_phase1(const float* __restrict__ k, const float* __restrict__ v,
                    float* __restrict__ ws)
{
  __shared__ float kt[DD*SMALL_STRIDE];   // kt[d][n], 32 x (32+2 pad)
  __shared__ float vt[DD*SMALL_STRIDE];   // vt[e][n]

  const int head = blockIdx.x;
  const int split = blockIdx.y;
  const int tid  = threadIdx.x;
  const int wave = tid >> 5, lane = tid & 31;
  const int half = lane >> 4, r = lane & 15;

  const float* kh = k + (size_t)head * NN * DD;
  const float* vh = v + (size_t)head * NN * DD;
  float* Pbase = ws + (size_t)(head * NSPLIT + split) * STAT_F32;

  v8f accS2[4][2][2] = {};      // 4 d1-segments per wave, 2x2 16x16 tiles
  v8f accX[2][2]     = {};      // wave0: S1, wave1: G
  float skacc = 0.f, svacc = 0.f;

  const int n0beg = split * NCHUNK;
  for (int n0 = n0beg; n0 < n0beg + NCHUNK; n0 += 32) {
    // ---- load + transpose the 32x32 k and v chunks into LDS ----
    #pragma unroll
    for (int i = 0; i < 4; ++i) {
      int lin = tid + 256*i;
      int n = lin >> 5, d = lin & 31;
      kt[d*SMALL_STRIDE + n] = kh[(size_t)(n0+n)*DD + d];
      vt[d*SMALL_STRIDE + n] = vh[(size_t)(n0+n)*DD + d];
    }
    if (n0 + 32 < n0beg + NCHUNK) {      // -> global_prefetch_b8 (next chunk)
      __builtin_prefetch(kh + (size_t)(n0+32)*DD + tid*4, 0, 1);
      __builtin_prefetch(vh + (size_t)(n0+32)*DD + tid*4, 0, 1);
    }
    __syncthreads();

    // ---- A tiles: k^T in bf16 A-layout (this code's contiguous k order, as in B: half0 K=0..15,
    //      half1 K=16..31; row M = lane&15) ----
    v16bf At[2];
    #pragma unroll
    for (int rt = 0; rt < 2; ++rt) {
      const int d2 = rt*16 + r;
      #pragma unroll
      for (int i = 0; i < 16; ++i) {
        const int nn = half*16 + i;
        At[rt][i] = (__bf16)kt[d2*SMALL_STRIDE + nn];
      }
    }

    // ---- extra segment: wave0 -> S1 (B = v), wave1 -> G (B = k) ----
    if (wave < 2) {
      const float* srcT = (wave == 0) ? vt : kt;
      const v16bf b0 = buildB(srcT, r,      half);
      const v16bf b1 = buildB(srcT, 16 + r, half);
      accX[0][0] = wmma_bf16(At[0], b0, accX[0][0]);
      accX[0][1] = wmma_bf16(At[0], b1, accX[0][1]);
      accX[1][0] = wmma_bf16(At[1], b0, accX[1][0]);
      accX[1][1] = wmma_bf16(At[1], b1, accX[1][1]);
    }

    // ---- 4 S2 segments per wave (static unroll -> acc stays in VGPRs) ----
    #pragma unroll
    for (int j = 0; j < 4; ++j) {
      const int d1 = wave*4 + j;
      const float* wRow = kt + d1*SMALL_STRIDE;
      const v16bf b0 = buildBw(vt, wRow, r,      half);
      const v16bf b1 = buildBw(vt, wRow, 16 + r, half);
      accS2[j][0][0] = wmma_bf16(At[0], b0, accS2[j][0][0]);
      accS2[j][0][1] = wmma_bf16(At[0], b1, accS2[j][0][1]);
      accS2[j][1][0] = wmma_bf16(At[1], b0, accS2[j][1][0]);
      accS2[j][1][1] = wmma_bf16(At[1], b1, accS2[j][1][1]);
    }

    if (wave == 2) {
      #pragma unroll
      for (int n = 0; n < 32; ++n) skacc += kt[lane*SMALL_STRIDE + n];
    }
    if (wave == 3) {
      #pragma unroll
      for (int n = 0; n < 32; ++n) svacc += vt[lane*SMALL_STRIDE + n];
    }
    __syncthreads();
  }

  // ---- write private partial image (covers every element; no init needed) ----
  #pragma unroll
  for (int j = 0; j < 4; ++j) {
    float* base = Pbase + S2_OFF + (wave*4 + j)*DD*DD;
    #pragma unroll
    for (int rt = 0; rt < 2; ++rt) {
      #pragma unroll
      for (int t = 0; t < 2; ++t) {
        #pragma unroll
        for (int i = 0; i < 8; ++i) {       // C layout: VGPR i -> M = i + 8*half
          const int m   = rt*16 + i + 8*half;
          const int col = t*16 + r;
          base[m*DD + col] = accS2[j][rt][t][i];
        }
      }
    }
  }
  if (wave < 2) {
    float* base = Pbase + ((wave == 0) ? S1_OFF : G_OFF);
    #pragma unroll
    for (int rt = 0; rt < 2; ++rt) {
      #pragma unroll
      for (int t = 0; t < 2; ++t) {
        #pragma unroll
        for (int i = 0; i < 8; ++i) {
          const int m   = rt*16 + i + 8*half;
          const int col = t*16 + r;
          base[m*DD + col] = accX[rt][t][i];
        }
      }
    }
  }
  if (wave == 2) Pbase[SK_OFF + lane] = skacc;
  if (wave == 3) Pbase[SV_OFF + lane] = svacc;
}

// ---------------------------------------------------------------------------
// Phase 1b: sum the NSPLIT partials; emit pre-scaled, transposed bf16
// B-operand images (column-major, padded strides) + f32 sv/sk.
// ---------------------------------------------------------------------------
__global__ __launch_bounds__(256)
void fastmax_convert(float* __restrict__ ws)
{
  const int head = blockIdx.x;
  const int tid = threadIdx.x;
  const float* P = ws + (size_t)head * NSPLIT * STAT_F32;
  char* regb = (char*)ws + REGB_OFF + (size_t)head * REGB_HEAD_BYTES;
  __bf16* s2t = (__bf16*)(regb + S2T_BYTE_OFF);
  __bf16* s1t = (__bf16*)(regb + S1T_BYTE_OFF);
  __bf16* gt  = (__bf16*)(regb + GT_BYTE_OFF);
  float*  svb = (float*)(regb + SVB_BYTE_OFF);
  float*  skb = (float*)(regb + SKB_BYTE_OFF);

  for (int idx = tid; idx < DD2*DD; idx += 256) {      // s2t[e][f] = A2*S2[f][e]
    const int e = idx >> 10, f = idx & (DD2-1);
    float s = 0.f;
    #pragma unroll
    for (int p = 0; p < NSPLIT; ++p) s += P[p*STAT_F32 + S2_OFF + f*DD + e];
    s2t[e*S2T_STRIDE + f] = (__bf16)(kA2 * s);
  }
  for (int idx = tid; idx < DD2; idx += 256) {         // s1t/gt column-major
    const int e = idx >> 5, d = idx & 31;
    float s1 = 0.f, g = 0.f;
    #pragma unroll
    for (int p = 0; p < NSPLIT; ++p) {
      s1 += P[p*STAT_F32 + S1_OFF + d*DD + e];
      g  += P[p*STAT_F32 + G_OFF  + d*DD + e];
    }
    s1t[e*SMALL_STRIDE + d] = (__bf16)(kA1 * s1);
    gt [e*SMALL_STRIDE + d] = (__bf16)(kA2 * g);
  }
  if (tid < DD) {
    float sv = 0.f, sk = 0.f;
    #pragma unroll
    for (int p = 0; p < NSPLIT; ++p) {
      sv += P[p*STAT_F32 + SV_OFF + tid];
      sk += P[p*STAT_F32 + SK_OFF + tid];
    }
    svb[tid] = sv;
    skb[tid] = kA1 * sk;
  }
}

// ---------------------------------------------------------------------------
// Phase 2: map over N.  Per 16-row wave-tile:
//   C  = (q@S1) + sum_d1 (q[:,d1].*q) @ S2block_d1      (66 WMMA)
//   T  = q @ G  (Gram == reshape(sk2))                  ( 2 WMMA)
//   den = a0*N + rowsum(q .* (a1*sk + T))  via shfl_xor reduction
//   out = (a0*sv + C) * (1/den)
// LDS staging via async global->LDS DMA (ASYNCcnt), no VGPR round-trip.
// ---------------------------------------------------------------------------
#define P2_ROWS    128
#define P2_THREADS 256
#define QS_STRIDE  33                                   // f32, bank skew
#define LDS_QS_OFF REGB_HEAD_BYTES
#define P2_LDS_BYTES (REGB_HEAD_BYTES + P2_ROWS*QS_STRIDE*4)   // 87168 B

__global__ __launch_bounds__(P2_THREADS)
void fastmax_phase2(const float* __restrict__ q, const float* __restrict__ ws,
                    float* __restrict__ out)
{
  extern __shared__ char smem[];
  const int head   = blockIdx.y;
  const int rowBlk = blockIdx.x;
  const int tid  = threadIdx.x;
  const int wave = tid >> 5, lane = tid & 31;
  const int half = lane >> 4, r = lane & 15;

  const __bf16* s2l = (const __bf16*)(smem + S2T_BYTE_OFF);
  const __bf16* s1l = (const __bf16*)(smem + S1T_BYTE_OFF);
  const __bf16* gl  = (const __bf16*)(smem + GT_BYTE_OFF);
  const float*  svl = (const float*)(smem + SVB_BYTE_OFF);
  const float*  skl = (const float*)(smem + SKB_BYTE_OFF);
  float* qs = (float*)(smem + LDS_QS_OFF);

  const float* qh = q + (size_t)head*NN*DD + (size_t)rowBlk*P2_ROWS*DD;

  // ---- async DMA stage: per-head bf16 image (LDS layout == global layout)
  //      and the q row tile (128B rows -> 132B-strided LDS image) ----
  {
    // dynamic LDS starts after static allocations (none here)
    const unsigned ldsBase = (unsigned)__builtin_amdgcn_groupstaticsize();
    const char* src = (const char*)ws + REGB_OFF + (size_t)head*REGB_HEAD_BYTES;
    for (int i = tid; i < REGB_HEAD_BYTES/16; i += P2_THREADS)
      async_ld_b128(ldsBase + (unsigned)i*16u,
                    (unsigned long long)(src + (size_t)i*16));
    for (int i = tid; i < P2_ROWS*8; i += P2_THREADS) {
      const int mr = i >> 3, part = i & 7;             // 8 x 16B per q row
      async_ld_b128(ldsBase + LDS_QS_OFF + (unsigned)(mr*QS_STRIDE*4 + part*16),
                    (unsigned long long)((const char*)qh + (size_t)(mr*DD + part*4)*4));
    }
    wait_asynccnt0();          // this wave's DMA done (LDS written)
  }
  __syncthreads();             // all waves' DMA done

  const int mloc0 = wave*16;
  const int mlocA = mloc0 + r;                // A row of this lane
  float qd2[16]; v16bf qA;
  #pragma unroll
  for (int i = 0; i < 16; ++i) {              // A K-slot -> d2 index
    const int d2 = half*16 + i;
    qd2[i] = qs[mlocA*QS_STRIDE + d2];
    qA[i] = (__bf16)qd2[i];
  }

  v8f C0 = {}, C1 = {}, T0 = {}, T1 = {};

  // S1 term (a1 folded into s1l) and denominator GEMM (a2 folded into gl)
  {
    v16bf b0, b1, g0, g1;
    const __bf16* p0 = s1l + (r     )*SMALL_STRIDE + half*16;
    const __bf16* p1 = s1l + (16 + r)*SMALL_STRIDE + half*16;
    const __bf16* q0 = gl  + (r     )*SMALL_STRIDE + half*16;
    const __bf16* q1 = gl  + (16 + r)*SMALL_STRIDE + half*16;
    #pragma unroll
    for (int i = 0; i < 16; ++i) { b0[i]=p0[i]; b1[i]=p1[i]; g0[i]=q0[i]; g1[i]=q1[i]; }
    C0 = wmma_bf16(qA, b0, C0);  C1 = wmma_bf16(qA, b1, C1);
    T0 = wmma_bf16(qA, g0, T0);  T1 = wmma_bf16(qA, g1, T1);
  }

  // main loop: C += (q[:,d1].*q) @ S2block_d1   (a2 folded into s2l)
  #pragma unroll 2
  for (int d1 = 0; d1 < DD; ++d1) {
    const float w = qs[mlocA*QS_STRIDE + d1];
    v16bf a;
    #pragma unroll
    for (int i = 0; i < 16; ++i) a[i] = (__bf16)(w * qd2[i]);
    v16bf b0, b1;
    const __bf16* p0 = s2l + (size_t)(r     )*S2T_STRIDE + d1*32 + half*16;
    const __bf16* p1 = s2l + (size_t)(16 + r)*S2T_STRIDE + d1*32 + half*16;
    #pragma unroll
    for (int i = 0; i < 16; ++i) { b0[i]=p0[i]; b1[i]=p1[i]; }
    C0 = wmma_bf16(a, b0, C0);
    C1 = wmma_bf16(a, b1, C1);
  }

  // denominator per row (shfl_xor tree within 16-lane halves) + fused store
  float* outh = out + (size_t)head*NN*DD;
  #pragma unroll
  for (int j = 0; j < 8; ++j) {
    const int mloc = mloc0 + j + 8*half;                 // C-layout row
    const float u0 = skl[r]      + T0[j];
    const float u1 = skl[r + 16] + T1[j];
    float partial = qs[mloc*QS_STRIDE + r]*u0 + qs[mloc*QS_STRIDE + r + 16]*u1;
    partial += __shfl_xor(partial, 1, 32);
    partial += __shfl_xor(partial, 2, 32);
    partial += __shfl_xor(partial, 4, 32);
    partial += __shfl_xor(partial, 8, 32);
    const float rinv = 1.0f / (kA0*(float)NN + partial);
    const size_t row = (size_t)rowBlk*P2_ROWS + mloc;
    outh[row*DD + r]      = (kA0*svl[r]      + C0[j]) * rinv;
    outh[row*DD + r + 16] = (kA0*svl[r + 16] + C1[j]) * rinv;
  }
}

// ---------------------------------------------------------------------------
extern "C" void kernel_launch(void* const* d_in, const int* in_sizes, int n_in,
                              void* d_out, int out_size, void* d_ws, size_t ws_size,
                              hipStream_t stream)
{
  (void)in_sizes; (void)n_in; (void)out_size; (void)ws_size;   // ws need ~19.2 MiB
  const float* q = (const float*)d_in[0];
  const float* k = (const float*)d_in[1];
  const float* v = (const float*)d_in[2];
  float* out = (float*)d_out;
  float* ws  = (float*)d_ws;

  fastmax_phase1<<<dim3(BHn, NSPLIT), 256, 0, stream>>>(k, v, ws);
  fastmax_convert<<<dim3(BHn), 256, 0, stream>>>(ws);
  fastmax_phase2<<<dim3(NN/P2_ROWS, BHn), P2_THREADS, P2_LDS_BYTES, stream>>>(q, ws, out);
}
